// MultiHeadSelfAttention_32667521254198
// MI455X (gfx1250) — matrix operand order repaired and checked by reading, not compiled, not run on hardware
//
#include <hip/hip_runtime.h>
#include <hip/hip_bf16.h>

// ---------------------------------------------------------------------------
// MHA with RoPE for MI455X (gfx1250), bf16 WMMA with f32 accumulation.
// B=4, S=2048, D=1024, H=16, dk=64.
// ---------------------------------------------------------------------------

typedef __attribute__((ext_vector_type(16))) __bf16 v16bf;
typedef __attribute__((ext_vector_type(8)))  __bf16 v8bf;
typedef __attribute__((ext_vector_type(8)))  float  v8f;
typedef int v4i_ __attribute__((vector_size(16)));   // type the async-LDS builtin expects

static constexpr int BATCH = 4;
static constexpr int SEQ   = 2048;
static constexpr int DM    = 1024;
static constexpr int NH    = 16;
static constexpr int DK    = 64;

#if __has_builtin(__builtin_amdgcn_global_load_async_to_lds_b128)
#define HAVE_ASYNC_LDS 1
#else
#define HAVE_ASYNC_LDS 0
#endif

// ---- bf16 <-> f32 helpers (bit-level, round-to-nearest-even) --------------
__device__ inline __bf16 f2bf(float f) {
    union { float f; unsigned u; } v; v.f = f;
    unsigned r = (v.u + 0x7FFFu + ((v.u >> 16) & 1u)) >> 16;
    union { unsigned short s; __bf16 b; } o; o.s = (unsigned short)r;
    return o.b;
}
__device__ inline float bf2f(__bf16 b) {
    union { unsigned short s; __bf16 b; } i; i.b = b;
    union { unsigned u; float f; } o; o.u = ((unsigned)i.s) << 16;
    return o.f;
}

// ---- WMMA wrapper ---------------------------------------------------------
__device__ inline v8f wmma_bf16(v16bf a, v16bf b, v8f c) {
    return __builtin_amdgcn_wmma_f32_16x16x32_bf16(
        /*neg_a=*/false, a, /*neg_b=*/false, b,
        /*c_mod=*/(short)0, c, /*reuse_a=*/false, /*reuse_b=*/false);
}

// ---- A-fragment loader ----------------------------------------------------
// 16-bit A 16x32 layout (ISA 7.12.2): lane L holds row M=L&15.
// half = L>>4.  elems i -> K = half*16 + i (contiguous order chosen here, B uses the same; not the ISA table order)
__device__ inline v16bf load_a_frag(const __bf16* base, int half) {
    v8bf lo = *(const v8bf*)(base + half * 16);
    v8bf hi = *(const v8bf*)(base + half * 16 + 8);
    return __builtin_shufflevector(lo, hi, 0,1,2,3,4,5,6,7,8,9,10,11,12,13,14,15);
}

// ---------------------------------------------------------------------------
// fp32 -> bf16 convert
// ---------------------------------------------------------------------------
__global__ void cvt_f32_bf16(const float* __restrict__ in, __bf16* __restrict__ out, size_t n) {
    size_t i = blockIdx.x * (size_t)blockDim.x + threadIdx.x;
    if (i < n) out[i] = f2bf(in[i]);
}

// ---------------------------------------------------------------------------
// GEMM: C[M,N] = A[M,K] * B[N,K]^T  (bf16 in, f32 accumulate)
// Register-blocked: each wave computes a 32(M) x 64(N) tile (8 WMMAs/K-step).
// grid = (N/64, M/128), block = 128 (4 waves stacked along M).
// ---------------------------------------------------------------------------
__global__ void wmma_gemm_bf16(const __bf16* __restrict__ A, const __bf16* __restrict__ B,
                               float* __restrict__ Cf, __bf16* __restrict__ Cb,
                               int M, int N, int K) {
    const int lane = threadIdx.x & 31;
    const int wave = threadIdx.x >> 5;           // 0..3
    const int n0   = blockIdx.x * 64;
    const int m0   = blockIdx.y * 128 + wave * 32;
    const int row  = lane & 15;
    const int half = lane >> 4;

    const __bf16* aRow0 = A + (size_t)(m0 + row) * K;
    const __bf16* aRow1 = A + (size_t)(m0 + 16 + row) * K;
    // B 32x16 fragment: lane holds column N=row; elems i -> K = half*16 + i
    const __bf16* bRow  = B + (size_t)(n0 + row) * K;

    v8f acc[2][4] = {};
    for (int k0 = 0; k0 < K; k0 += 32) {
        v16bf a0 = load_a_frag(aRow0 + k0, half);
        v16bf a1 = load_a_frag(aRow1 + k0, half);
#pragma unroll
        for (int nc = 0; nc < 4; ++nc) {
            v16bf b = *(const v16bf*)(bRow + (size_t)nc * 16 * K + k0 + half * 16);
            acc[0][nc] = wmma_bf16(a0, b, acc[0][nc]);
            acc[1][nc] = wmma_bf16(a1, b, acc[1][nc]);
        }
    }

    // C layout: VGPR r -> row M = r + half*8 ; col N = lane&15
#pragma unroll
    for (int mi = 0; mi < 2; ++mi) {
#pragma unroll
        for (int nc = 0; nc < 4; ++nc) {
#pragma unroll
            for (int r = 0; r < 8; ++r) {
                int m = m0 + mi * 16 + r + half * 8;
                int n = n0 + nc * 16 + row;
                if (Cf) Cf[(size_t)m * N + n] = acc[mi][nc][r];
                if (Cb) Cb[(size_t)m * N + n] = f2bf(acc[mi][nc][r]);
            }
        }
    }
}

// ---------------------------------------------------------------------------
// RoPE applied in-place to Q and K (bf16, layout [B,S,H*dk]).
// ---------------------------------------------------------------------------
__global__ void rope_kernel(__bf16* __restrict__ q, __bf16* __restrict__ k,
                            const int* __restrict__ use_rope, size_t total_pairs) {
    if (*use_rope == 0) return;
    size_t idx = blockIdx.x * (size_t)blockDim.x + threadIdx.x;
    if (idx >= total_pairs) return;
    const int pairs_per_row = DM / 2;             // 512 pairs per (b,s)
    size_t rowi = idx / pairs_per_row;            // b*S + s
    int p  = (int)(idx % pairs_per_row);          // p = h*(dk/2) + j
    int s  = (int)(rowi % SEQ);
    int j  = p & (DK / 2 - 1);                    // j in [0, 32)
    float inv = __powf(10000.0f, -2.0f * (float)j / (float)DK);
    float ang = (float)s * inv;
    float cs = __cosf(ang), sn = __sinf(ang);
    size_t base = rowi * DM + (size_t)p * 2;      // p*2 == h*dk + 2j

    float qe = bf2f(q[base]), qo = bf2f(q[base + 1]);
    q[base]     = f2bf(qe * cs - qo * sn);
    q[base + 1] = f2bf(qo * cs + qe * sn);

    float ke = bf2f(k[base]), ko = bf2f(k[base + 1]);
    k[base]     = f2bf(ke * cs - ko * sn);
    k[base + 1] = f2bf(ko * cs + ke * sn);
}

// ---------------------------------------------------------------------------
// Flash attention (causal, online softmax). One wave per 16-query tile.
// grid = (S/16, B*H), block = 32 (single-wave WG: barriers are S_NOP).
// Q,K,V,O layout: [B, S, H*dk] bf16.
// ---------------------------------------------------------------------------
__global__ void flash_attn(const __bf16* __restrict__ Q, const __bf16* __restrict__ Km,
                           const __bf16* __restrict__ V, __bf16* __restrict__ O) {
    __shared__ __bf16 lds_p[16 * 32];   // P tile, row-major 16x32
    __shared__ __bf16 lds_v[32 * 64];   // V tile, row-major 32x64

    const int lane = threadIdx.x & 31;
    const int row  = lane & 15;
    const int half = lane >> 4;
    const int m0   = blockIdx.x * 16;
    const int bh   = blockIdx.y;
    const int b    = bh / NH, h = bh % NH;
    const size_t baseBH = ((size_t)b * SEQ) * DM + (size_t)h * DK;

    // Q fragments (16x64 split into two 16x32 A fragments)
    const __bf16* qrow = Q + baseBH + (size_t)(m0 + row) * DM;
    v16bf aq0 = load_a_frag(qrow + 0,  half);
    v16bf aq1 = load_a_frag(qrow + 32, half);

    v8f acc0 = {}, acc1 = {}, acc2 = {}, acc3 = {};
    float mrow[8], lrow[8];
#pragma unroll
    for (int r = 0; r < 8; ++r) { mrow[r] = -3.0e38f; lrow[r] = 0.0f; }

    const float scale = 0.125f;   // 1/sqrt(64)

    for (int j0 = 0; j0 <= m0 + 15; j0 += 32) {
        // ---- stage V tile (32 keys x 64 dims = 4KB) into LDS --------------
        // 256 x 16B chunks; 32 lanes x 8 iterations.
#pragma unroll
        for (int it = 0; it < 8; ++it) {
            int cid = it * 32 + lane;
            int vr  = cid >> 3;          // key row 0..31
            int c16 = cid & 7;           // 16B chunk within row
            const __bf16* gsrc = V + baseBH + (size_t)(j0 + vr) * DM + c16 * 8;
            __bf16* ldst = &lds_v[vr * 64 + c16 * 8];
#if HAVE_ASYNC_LDS
            __builtin_amdgcn_global_load_async_to_lds_b128(
                (v4i_*)gsrc, (v4i_*)ldst, /*offset=*/0, /*cpol=*/0);
#else
            *(uint4*)ldst = *(const uint4*)gsrc;
#endif
        }

        // ---- S tile = Q(16x64) * K_tile^T(64x32), two 16-col halves -------
        const __bf16* kr0 = Km + baseBH + (size_t)(j0 + row) * DM;
        const __bf16* kr1 = Km + baseBH + (size_t)(j0 + 16 + row) * DM;
        v16bf bk00 = *(const v16bf*)(kr0 + half * 16);        // keys j0..+15, d 0..31
        v16bf bk01 = *(const v16bf*)(kr0 + 32 + half * 16);   // keys j0..+15, d 32..63
        v16bf bk10 = *(const v16bf*)(kr1 + half * 16);
        v16bf bk11 = *(const v16bf*)(kr1 + 32 + half * 16);

        v8f s0 = {}, s1 = {};
        s0 = wmma_bf16(aq0, bk00, s0);
        s0 = wmma_bf16(aq1, bk01, s0);
        s1 = wmma_bf16(aq0, bk10, s1);
        s1 = wmma_bf16(aq1, bk11, s1);

        // ---- causal mask + online softmax ---------------------------------
#pragma unroll
        for (int r = 0; r < 8; ++r) {
            int mq  = m0 + r + half * 8;           // global query index
            int nk0 = j0 + row;                    // key index, first half
            int nk1 = j0 + 16 + row;               // key index, second half
            float v0 = s0[r] * scale; if (nk0 > mq) v0 = -3.0e38f;
            float v1 = s1[r] * scale; if (nk1 > mq) v1 = -3.0e38f;

            float rm = fmaxf(v0, v1);
            rm = fmaxf(rm, __shfl_xor(rm, 1));
            rm = fmaxf(rm, __shfl_xor(rm, 2));
            rm = fmaxf(rm, __shfl_xor(rm, 4));
            rm = fmaxf(rm, __shfl_xor(rm, 8));
            float mn = fmaxf(mrow[r], rm);

            float p0 = __expf(v0 - mn);
            float p1 = __expf(v1 - mn);
            float rs = p0 + p1;
            rs += __shfl_xor(rs, 1);
            rs += __shfl_xor(rs, 2);
            rs += __shfl_xor(rs, 4);
            rs += __shfl_xor(rs, 8);

            float alpha = __expf(mrow[r] - mn);
            lrow[r] = lrow[r] * alpha + rs;
            mrow[r] = mn;
            acc0[r] *= alpha; acc1[r] *= alpha; acc2[r] *= alpha; acc3[r] *= alpha;

            int M = r + half * 8;                  // local row in tile
            lds_p[M * 32 + row]      = f2bf(p0);
            lds_p[M * 32 + 16 + row] = f2bf(p1);
        }
        __syncthreads();
#if HAVE_ASYNC_LDS
        asm volatile("s_wait_asynccnt 0" ::: "memory");
#endif

        // ---- re-stripe P from C-layout (via LDS) into A-fragment ----------
        v16bf ap;
        {
            const __bf16* pr = lds_p + row * 32;
            v8bf lo = *(const v8bf*)(pr + half * 16);
            v8bf hi = *(const v8bf*)(pr + half * 16 + 8);
            ap = __builtin_shufflevector(lo, hi, 0,1,2,3,4,5,6,7,8,9,10,11,12,13,14,15);
        }

        // ---- V B-fragments from LDS: elem i -> key = half*16+i, d = nc*16+row
        v16bf bv0, bv1, bv2, bv3;
#pragma unroll
        for (int i = 0; i < 16; ++i) {
            const __bf16* vr = lds_v + (half * 16 + i) * 64 + row;
            bv0[i] = vr[0];
            bv1[i] = vr[16];
            bv2[i] = vr[32];
            bv3[i] = vr[48];
        }

        acc0 = wmma_bf16(ap, bv0, acc0);
        acc1 = wmma_bf16(ap, bv1, acc1);
        acc2 = wmma_bf16(ap, bv2, acc2);
        acc3 = wmma_bf16(ap, bv3, acc3);
        __syncthreads();
    }

    // ---- normalize and write O -------------------------------------------
#pragma unroll
    for (int r = 0; r < 8; ++r) {
        float inv = 1.0f / lrow[r];
        size_t obase = baseBH + (size_t)(m0 + r + half * 8) * DM;
        O[obase + row]      = f2bf(acc0[r] * inv);
        O[obase + 16 + row] = f2bf(acc1[r] * inv);
        O[obase + 32 + row] = f2bf(acc2[r] * inv);
        O[obase + 48 + row] = f2bf(acc3[r] * inv);
    }
}

// ---------------------------------------------------------------------------
// Launch
// ---------------------------------------------------------------------------
extern "C" void kernel_launch(void* const* d_in, const int* in_sizes, int n_in,
                              void* d_out, int out_size, void* d_ws, size_t ws_size,
                              hipStream_t stream) {
    const float* x  = (const float*)d_in[0];
    const float* Wq = (const float*)d_in[1];
    const float* Wk = (const float*)d_in[2];
    const float* Wv = (const float*)d_in[3];
    const float* Wo = (const float*)d_in[4];
    const int* use_rope = (const int*)d_in[5];

    const int M = BATCH * SEQ;          // 8192
    const size_t MD = (size_t)M * DM;   // 8,388,608 elements
    const size_t DD = (size_t)DM * DM;  // 1,048,576 elements

    char* p = (char*)d_ws;
    __bf16* xb  = (__bf16*)p; p += MD * 2;
    __bf16* wqb = (__bf16*)p; p += DD * 2;
    __bf16* wkb = (__bf16*)p; p += DD * 2;
    __bf16* wvb = (__bf16*)p; p += DD * 2;
    __bf16* wob = (__bf16*)p; p += DD * 2;
    __bf16* qb  = (__bf16*)p; p += MD * 2;
    __bf16* kb  = (__bf16*)p; p += MD * 2;
    __bf16* vb  = (__bf16*)p; p += MD * 2;
    __bf16* ob  = (__bf16*)p; p += MD * 2;

    // fp32 -> bf16 converts
    cvt_f32_bf16<<<(unsigned)((MD + 255) / 256), 256, 0, stream>>>(x, xb, MD);
    cvt_f32_bf16<<<(unsigned)((DD + 255) / 256), 256, 0, stream>>>(Wq, wqb, DD);
    cvt_f32_bf16<<<(unsigned)((DD + 255) / 256), 256, 0, stream>>>(Wk, wkb, DD);
    cvt_f32_bf16<<<(unsigned)((DD + 255) / 256), 256, 0, stream>>>(Wv, wvb, DD);
    cvt_f32_bf16<<<(unsigned)((DD + 255) / 256), 256, 0, stream>>>(Wo, wob, DD);

    // QKV projections: [M,DM] = x[M,DM] * W[DM,DM]^T
    dim3 gg(DM / 64, M / 128);
    wmma_gemm_bf16<<<gg, 128, 0, stream>>>(xb, wqb, nullptr, qb, M, DM, DM);
    wmma_gemm_bf16<<<gg, 128, 0, stream>>>(xb, wkb, nullptr, kb, M, DM, DM);
    wmma_gemm_bf16<<<gg, 128, 0, stream>>>(xb, wvb, nullptr, vb, M, DM, DM);

    // RoPE on Q,K
    size_t pairs = MD / 2;
    rope_kernel<<<(unsigned)((pairs + 255) / 256), 256, 0, stream>>>(qb, kb, use_rope, pairs);

    // Causal flash attention
    dim3 ga(SEQ / 16, BATCH * NH);
    flash_attn<<<ga, 32, 0, stream>>>(qb, kb, vb, ob);

    // Output projection -> fp32 d_out
    wmma_gemm_bf16<<<gg, 128, 0, stream>>>(ob, wob, (float*)d_out, nullptr, M, DM, DM);
}
